// Transformer_19267223290388
// MI455X (gfx1250) — matrix operand order repaired and checked by reading, not compiled, not run on hardware
//
#include <hip/hip_runtime.h>
#include <hip/hip_bf16.h>

// ---------------------------------------------------------------------------
// Types for gfx1250 WMMA
// ---------------------------------------------------------------------------
typedef __attribute__((ext_vector_type(16))) __bf16 v16bf;
typedef __attribute__((ext_vector_type(8)))  float  v8f;

union Frag {
    v16bf v;
    uint4 q[2];
};

__device__ __forceinline__ unsigned short f2bf(float f) {
    unsigned u = __float_as_uint(f);
    u += 0x7FFFu + ((u >> 16) & 1u);          // round-to-nearest-even
    return (unsigned short)(u >> 16);
}
__device__ __forceinline__ unsigned pack2bf(float a, float b) {
    return (unsigned)f2bf(a) | ((unsigned)f2bf(b) << 16);
}

__device__ __forceinline__ v8f wmma_bf16(const Frag& a, const Frag& b, v8f c) {
    return __builtin_amdgcn_wmma_f32_16x16x32_bf16(
        false, a.v, false, b.v, (short)0, c, false, false);
}

// ---------------------------------------------------------------------------
// GEMM: C[M,N] = A[M,K] @ W[K,N] + bias (+ReLU | +residual)
// A,W,bias fp32; bf16 WMMA with fp32 accumulate.
// Block: 128 threads (4 waves), 64x64 C tile, each wave a 32x32 quadrant.
// EPI: 0 = bias, 1 = bias+relu, 2 = bias+residual
// ---------------------------------------------------------------------------
template <int EPI>
__global__ __launch_bounds__(128)
void gemm_kernel(const float* __restrict__ A, const float* __restrict__ W,
                 const float* __restrict__ bias, const float* __restrict__ Rsd,
                 float* __restrict__ C, int M, int N, int K) {
    __shared__ __align__(16) unsigned sA[64 * 16];  // [m][kpair] bf16x2, K-tile=32
    __shared__ __align__(16) unsigned sB[64 * 16];  // [n][kpair] (W transposed)

    const int t = threadIdx.x;
    const int wave = t >> 5, lane = t & 31;
    const int half = lane >> 4, lm = lane & 15;
    const int wm = wave >> 1, wn = wave & 1;
    const int m0 = blockIdx.y * 64, n0 = blockIdx.x * 64;

    v8f acc[2][2];
    acc[0][0] = {}; acc[0][1] = {}; acc[1][0] = {}; acc[1][1] = {};

    for (int k0 = 0; k0 < K; k0 += 32) {
        __syncthreads();
        // A tile 64x32 fp32 -> bf16 LDS (row-major)
#pragma unroll
        for (int i = 0; i < 4; ++i) {
            int idx = t + i * 128;                 // 0..511 float4s
            int row = idx >> 3, kq = idx & 7;
            float4 f = *(const float4*)&A[(size_t)(m0 + row) * K + k0 + kq * 4];
            sA[row * 16 + kq * 2 + 0] = pack2bf(f.x, f.y);
            sA[row * 16 + kq * 2 + 1] = pack2bf(f.z, f.w);
        }
        // W tile 32x64 fp32 -> transposed bf16 LDS: sB[n][k]
#pragma unroll
        for (int i = 0; i < 8; ++i) {
            int idx = t + i * 128;                 // 0..1023
            int n = idx & 63, kp = idx >> 6;       // kp in 0..15
            const float* wp = &W[(size_t)(k0 + kp * 2) * N + n0 + n];
            sB[n * 16 + kp] = pack2bf(wp[0], wp[N]);
        }
        __syncthreads();

        const uint4* Aq = (const uint4*)sA;        // 4 uint4 per row
        const uint4* Bq = (const uint4*)sB;
        Frag a[2], b[2];
#pragma unroll
        for (int mt = 0; mt < 2; ++mt) {
            int row = wm * 32 + mt * 16 + lm;
            a[mt].q[0] = Aq[row * 4 + half];       // K = half*8 + 0..7
            a[mt].q[1] = Aq[row * 4 + 2 + half];   // K = 16 + half*8 + 0..7
        }
#pragma unroll
        for (int nt = 0; nt < 2; ++nt) {
            int col = wn * 32 + nt * 16 + lm;
            b[nt].q[0] = Bq[col * 4 + half];       // K = half*8 + 0..7
            b[nt].q[1] = Bq[col * 4 + 2 + half];
        }
#pragma unroll
        for (int mt = 0; mt < 2; ++mt)
#pragma unroll
            for (int nt = 0; nt < 2; ++nt)
                acc[mt][nt] = wmma_bf16(a[mt], b[nt], acc[mt][nt]);
    }

#pragma unroll
    for (int mt = 0; mt < 2; ++mt)
#pragma unroll
        for (int nt = 0; nt < 2; ++nt) {
            int col = n0 + wn * 32 + nt * 16 + lm;
            float bv = bias[col];
#pragma unroll
            for (int r = 0; r < 8; ++r) {
                int row = m0 + wm * 32 + mt * 16 + half * 8 + r;
                float vv = acc[mt][nt][r] + bv;
                if (EPI == 1) vv = fmaxf(vv, 0.f);
                if (EPI == 2) vv += Rsd[(size_t)row * N + col];
                C[(size_t)row * N + col] = vv;
            }
        }
}

// ---------------------------------------------------------------------------
// Flash-attention. Q,K,V: [B*Sq,D]/[B*Sk,D] fp32, head h uses cols h*64..h*64+63.
// Grid: (Sq/64, H, B), block 128 (4 waves); each wave owns 16 query rows.
// CAUSAL: tril mask; else maskv[b*Sk+key]==0 -> -1e9.
// ---------------------------------------------------------------------------
template <bool CAUSAL>
__global__ __launch_bounds__(128)
void attn_kernel(const float* __restrict__ Q, const float* __restrict__ Kg,
                 const float* __restrict__ Vg, const int* __restrict__ maskv,
                 float* __restrict__ O, int Sq, int Sk, int D) {
    __shared__ __align__(16) unsigned sK[32 * 32];        // [key][dkpair]  32x64 bf16
    __shared__ __align__(16) unsigned sVT[64 * 16];       // [dim][keypair] 64x32 bf16
    __shared__ __align__(16) unsigned short sP[4][16 * 32]; // per-wave P 16x32 bf16

    const int t = threadIdx.x, wave = t >> 5, lane = t & 31;
    const int half = lane >> 4, lm = lane & 15;
    const int b = blockIdx.z, h = blockIdx.y, qt = blockIdx.x;
    const int hb = h * 64;
    const int qbase = qt * 64 + wave * 16;

    // Q fragments (rows qbase..qbase+15, dk 0..63), pre-scaled by 1/sqrt(64)
    Frag qf[2];
    {
        const float* qp = &Q[(size_t)(b * Sq + qbase + lm) * D + hb];
#pragma unroll
        for (int c = 0; c < 2; ++c) {
            float4 f0 = *(const float4*)&qp[c * 32 + half * 8];
            float4 f1 = *(const float4*)&qp[c * 32 + half * 8 + 4];
            float4 f2 = *(const float4*)&qp[c * 32 + 16 + half * 8];
            float4 f3 = *(const float4*)&qp[c * 32 + 16 + half * 8 + 4];
            const float s = 0.125f;
            qf[c].q[0] = make_uint4(pack2bf(f0.x * s, f0.y * s), pack2bf(f0.z * s, f0.w * s),
                                    pack2bf(f1.x * s, f1.y * s), pack2bf(f1.z * s, f1.w * s));
            qf[c].q[1] = make_uint4(pack2bf(f2.x * s, f2.y * s), pack2bf(f2.z * s, f2.w * s),
                                    pack2bf(f3.x * s, f3.y * s), pack2bf(f3.z * s, f3.w * s));
        }
    }

    v8f o[4];
    o[0] = {}; o[1] = {}; o[2] = {}; o[3] = {};
    float rm[8], rs[8];
#pragma unroll
    for (int r = 0; r < 8; ++r) { rm[r] = -1e30f; rs[r] = 0.f; }

    int nkt = Sk >> 5;
    if (CAUSAL) { int cmax = (int)blockIdx.x * 2 + 2; if (cmax < nkt) nkt = cmax; }

    for (int kt = 0; kt < nkt; ++kt) {
        const int sk = kt * 32;
        __syncthreads();  // previous tile fully consumed
        // K tile 32 keys x 64 dk -> sK[key][dkpair]
#pragma unroll
        for (int i = 0; i < 8; ++i) {
            int idx = t + i * 128;
            int dp = idx & 31, key = idx >> 5;
            float2 f = *(const float2*)&Kg[(size_t)(b * Sk + sk + key) * D + hb + dp * 2];
            sK[key * 32 + dp] = pack2bf(f.x, f.y);
        }
        // V tile transposed -> sVT[dim][keypair]
#pragma unroll
        for (int i = 0; i < 8; ++i) {
            int idx = t + i * 128;
            int dim = idx & 63, kp = idx >> 6;
            const float* vp = &Vg[(size_t)(b * Sk + sk + kp * 2) * D + hb + dim];
            sVT[dim * 16 + kp] = pack2bf(vp[0], vp[D]);
        }
        __syncthreads();

        // scores: S = (Q/8) . K^T   (B-matrix = K rows, already the right layout)
        const uint4* Kq = (const uint4*)sK;        // 8 uint4 per key row
        v8f s[2];
#pragma unroll
        for (int nt = 0; nt < 2; ++nt) {
            int key = nt * 16 + lm;
            Frag kb0, kb1;
            kb0.q[0] = Kq[key * 8 + half];
            kb0.q[1] = Kq[key * 8 + 2 + half];
            kb1.q[0] = Kq[key * 8 + 4 + half];
            kb1.q[1] = Kq[key * 8 + 6 + half];
            v8f a = {};
            a = wmma_bf16(qf[0], kb0, a);
            a = wmma_bf16(qf[1], kb1, a);
            s[nt] = a;
        }

        // mask
        if (CAUSAL) {
#pragma unroll
            for (int nt = 0; nt < 2; ++nt) {
                int key = sk + nt * 16 + lm;
#pragma unroll
                for (int r = 0; r < 8; ++r) {
                    int qrow = qbase + half * 8 + r;
                    if (key > qrow) s[nt][r] = -1e9f;
                }
            }
        } else {
#pragma unroll
            for (int nt = 0; nt < 2; ++nt) {
                int key = sk + nt * 16 + lm;
                if (maskv[b * Sk + key] == 0) {
#pragma unroll
                    for (int r = 0; r < 8; ++r) s[nt][r] = -1e9f;
                }
            }
        }

        // online softmax update (row stats replicated across each 16-lane group)
        float alpha[8];
#pragma unroll
        for (int r = 0; r < 8; ++r) {
            float mx = fmaxf(s[0][r], s[1][r]);
#pragma unroll
            for (int off = 8; off > 0; off >>= 1) mx = fmaxf(mx, __shfl_xor(mx, off, 32));
            float mnew = fmaxf(rm[r], mx);
            alpha[r] = __expf(rm[r] - mnew);
            float p0 = __expf(s[0][r] - mnew);
            float p1 = __expf(s[1][r] - mnew);
            float ps = p0 + p1;
#pragma unroll
            for (int off = 8; off > 0; off >>= 1) ps += __shfl_xor(ps, off, 32);
            rs[r] = rs[r] * alpha[r] + ps;
            rm[r] = mnew;
            s[0][r] = p0; s[1][r] = p1;
        }
#pragma unroll
        for (int nt = 0; nt < 4; ++nt)
#pragma unroll
            for (int r = 0; r < 8; ++r) o[nt][r] *= alpha[r];

        // stage P (C-layout -> A-layout via per-wave LDS)
        unsigned short* pl = sP[wave];
#pragma unroll
        for (int nt = 0; nt < 2; ++nt)
#pragma unroll
            for (int r = 0; r < 8; ++r)
                pl[(half * 8 + r) * 32 + nt * 16 + lm] = f2bf(s[nt][r]);
        __syncthreads();

        // O += P @ V  (B-matrix = V^T from sVT)
        const uint4* Pq = (const uint4*)sP[wave];  // 4 uint4 per row
        Frag pf;
        pf.q[0] = Pq[lm * 4 + half];
        pf.q[1] = Pq[lm * 4 + 2 + half];
        const uint4* Vq = (const uint4*)sVT;
#pragma unroll
        for (int nt = 0; nt < 4; ++nt) {
            int dim = nt * 16 + lm;
            Frag vb;
            vb.q[0] = Vq[dim * 4 + half];
            vb.q[1] = Vq[dim * 4 + 2 + half];
            o[nt] = wmma_bf16(pf, vb, o[nt]);
        }
    }

    // finalize
    float inv[8];
#pragma unroll
    for (int r = 0; r < 8; ++r) inv[r] = 1.f / rs[r];
#pragma unroll
    for (int nt = 0; nt < 4; ++nt)
#pragma unroll
        for (int r = 0; r < 8; ++r) {
            int row = b * Sq + qbase + half * 8 + r;
            O[(size_t)row * D + hb + nt * 16 + lm] = o[nt][r] * inv[r];
        }
}

// ---------------------------------------------------------------------------
// Embedding * sqrt(D) + sinusoidal PE (computed analytically)
// ---------------------------------------------------------------------------
__global__ __launch_bounds__(256)
void embed_kernel(const int* __restrict__ tok, const float* __restrict__ emb,
                  float* __restrict__ out, int S, int D) {
    const int row = blockIdx.x;
    const int pos = row % S;
    const int tk = tok[row];
    for (int c = threadIdx.x; c < D; c += 256) {
        float ang = (float)pos * __expf(-(float)(c & ~1) * (9.210340372f / 512.f));
        float pe = (c & 1) ? __cosf(ang) : __sinf(ang);
        out[(size_t)row * D + c] = emb[(size_t)tk * D + c] * 22.62741699796952f + pe;
    }
}

// ---------------------------------------------------------------------------
// LayerNorm over D=512 (unbiased variance, ddof=1). One wave per row.
// ---------------------------------------------------------------------------
__global__ __launch_bounds__(128)
void ln_kernel(const float* __restrict__ X, const float* __restrict__ g,
               const float* __restrict__ bta, float* __restrict__ Y) {
    const int row = blockIdx.x * 4 + (threadIdx.x >> 5);
    const int lane = threadIdx.x & 31;
    const float* x = X + (size_t)row * 512;
    float vals[16];
    float s = 0.f;
#pragma unroll
    for (int i = 0; i < 16; ++i) { vals[i] = x[lane + i * 32]; s += vals[i]; }
#pragma unroll
    for (int off = 16; off > 0; off >>= 1) s += __shfl_xor(s, off, 32);
    float mean = s * (1.f / 512.f);
    float var = 0.f;
#pragma unroll
    for (int i = 0; i < 16; ++i) { float d = vals[i] - mean; var += d * d; }
#pragma unroll
    for (int off = 16; off > 0; off >>= 1) var += __shfl_xor(var, off, 32);
    var *= (1.f / 511.f);
    float inv = rsqrtf(var + 1e-6f);
#pragma unroll
    for (int i = 0; i < 16; ++i) {
        int c = lane + i * 32;
        Y[(size_t)row * 512 + c] = g[c] * (vals[i] - mean) * inv + bta[c];
    }
}

// ---------------------------------------------------------------------------
// In-place log_softmax over rows of width V (V=32000)
// ---------------------------------------------------------------------------
__global__ __launch_bounds__(256)
void lsm_kernel(float* __restrict__ X, int V) {
    __shared__ float red[256];
    const int tid = threadIdx.x;
    float* x = X + (size_t)blockIdx.x * V;

    float mx = -3.4e38f;
    for (int c = tid; c < V; c += 256) mx = fmaxf(mx, x[c]);
    red[tid] = mx; __syncthreads();
    for (int off = 128; off > 0; off >>= 1) {
        if (tid < off) red[tid] = fmaxf(red[tid], red[tid + off]);
        __syncthreads();
    }
    float rmax = red[0];
    __syncthreads();

    float s = 0.f;
    for (int c = tid; c < V; c += 256) s += __expf(x[c] - rmax);
    red[tid] = s; __syncthreads();
    for (int off = 128; off > 0; off >>= 1) {
        if (tid < off) red[tid] += red[tid + off];
        __syncthreads();
    }
    float lse = rmax + logf(red[0]);
    for (int c = tid; c < V; c += 256) x[c] = x[c] - lse;
}

// ---------------------------------------------------------------------------
// Host orchestration
// ---------------------------------------------------------------------------
extern "C" void kernel_launch(void* const* d_in, const int* in_sizes, int n_in,
                              void* d_out, int out_size, void* d_ws, size_t ws_size,
                              hipStream_t stream) {
    (void)in_sizes; (void)n_in; (void)out_size; (void)ws_size;

    constexpr int D = 512, H = 8, F = 2048, NL = 2, S = 1024, T = 1024;
    constexpr int ROWS = 2 * 1024;  // B*S == B*T

    const int*   src      = (const int*)d_in[0];
    const int*   tgt      = (const int*)d_in[1];
    const int*   src_mask = (const int*)d_in[2];   // [B,1,1,S] -> flat [B*S]
    const float* src_emb  = (const float*)d_in[4];
    const float* tgt_emb  = (const float*)d_in[5];
    const float* ea_w  = (const float*)d_in[6];
    const float* ea_b  = (const float*)d_in[7];
    const float* ef_w1 = (const float*)d_in[8];
    const float* ef_b1 = (const float*)d_in[9];
    const float* ef_w2 = (const float*)d_in[10];
    const float* ef_b2 = (const float*)d_in[11];
    const float* eln_g = (const float*)d_in[12];
    const float* eln_b = (const float*)d_in[13];
    const float* en_g  = (const float*)d_in[14];
    const float* en_b  = (const float*)d_in[15];
    const float* ds_w  = (const float*)d_in[16];
    const float* ds_b  = (const float*)d_in[17];
    const float* dc_w  = (const float*)d_in[18];
    const float* dc_b  = (const float*)d_in[19];
    const float* df_w1 = (const float*)d_in[20];
    const float* df_b1 = (const float*)d_in[21];
    const float* df_w2 = (const float*)d_in[22];
    const float* df_b2 = (const float*)d_in[23];
    const float* dln_g = (const float*)d_in[24];
    const float* dln_b = (const float*)d_in[25];
    const float* dn_g  = (const float*)d_in[26];
    const float* dn_b  = (const float*)d_in[27];
    const float* pw    = (const float*)d_in[28];
    const float* pb    = (const float*)d_in[29];
    float* out = (float*)d_out;

    // workspace layout (floats)
    float* ws  = (float*)d_ws;
    float* xe  = ws;                         // encoder activations [2048,512]
    float* yd  = xe  + (size_t)ROWS * D;     // decoder activations
    float* hbuf= yd  + (size_t)ROWS * D;     // LN output
    float* qb  = hbuf+ (size_t)ROWS * D;
    float* kb  = qb  + (size_t)ROWS * D;
    float* vb  = kb  + (size_t)ROWS * D;
    float* ab  = vb  + (size_t)ROWS * D;     // attention output (concat heads)
    float* enc = ab  + (size_t)ROWS * D;     // encoder final
    float* fb  = enc + (size_t)ROWS * D;     // FFN mid [2048,2048]

    auto gemm = [&](const float* A, const float* W, const float* bias,
                    const float* rsd, float* C, int M, int N, int K, int epi) {
        dim3 grid(N / 64, M / 64);
        if (epi == 0)
            gemm_kernel<0><<<grid, 128, 0, stream>>>(A, W, bias, nullptr, C, M, N, K);
        else if (epi == 1)
            gemm_kernel<1><<<grid, 128, 0, stream>>>(A, W, bias, nullptr, C, M, N, K);
        else
            gemm_kernel<2><<<grid, 128, 0, stream>>>(A, W, bias, rsd, C, M, N, K);
    };

    const dim3 agrid(S / 64, H, 2);

    // ---------------- encoder ----------------
    embed_kernel<<<ROWS, 256, 0, stream>>>(src, src_emb, xe, S, D);
    for (int i = 0; i < NL; ++i) {
        const float* aw = ea_w + (size_t)i * 4 * D * D;
        const float* abp = ea_b + (size_t)i * 4 * D;
        ln_kernel<<<ROWS / 4, 128, 0, stream>>>(xe, eln_g + (size_t)(i * 2) * D,
                                                eln_b + (size_t)(i * 2) * D, hbuf);
        gemm(hbuf, aw + 0 * (size_t)D * D, abp + 0 * D, nullptr, qb, ROWS, D, D, 0);
        gemm(hbuf, aw + 1 * (size_t)D * D, abp + 1 * D, nullptr, kb, ROWS, D, D, 0);
        gemm(hbuf, aw + 2 * (size_t)D * D, abp + 2 * D, nullptr, vb, ROWS, D, D, 0);
        attn_kernel<false><<<agrid, 128, 0, stream>>>(qb, kb, vb, src_mask, ab, S, S, D);
        gemm(ab, aw + 3 * (size_t)D * D, abp + 3 * D, xe, xe, ROWS, D, D, 2);
        ln_kernel<<<ROWS / 4, 128, 0, stream>>>(xe, eln_g + (size_t)(i * 2 + 1) * D,
                                                eln_b + (size_t)(i * 2 + 1) * D, hbuf);
        gemm(hbuf, ef_w1 + (size_t)i * D * F, ef_b1 + (size_t)i * F, nullptr, fb, ROWS, F, D, 1);
        gemm(fb, ef_w2 + (size_t)i * F * D, ef_b2 + (size_t)i * D, xe, xe, ROWS, D, F, 2);
    }
    ln_kernel<<<ROWS / 4, 128, 0, stream>>>(xe, en_g, en_b, enc);

    // ---------------- decoder ----------------
    embed_kernel<<<ROWS, 256, 0, stream>>>(tgt, tgt_emb, yd, T, D);
    for (int i = 0; i < NL; ++i) {
        const float* sw = ds_w + (size_t)i * 4 * D * D;
        const float* sb = ds_b + (size_t)i * 4 * D;
        const float* cw = dc_w + (size_t)i * 4 * D * D;
        const float* cb = dc_b + (size_t)i * 4 * D;
        // masked self-attention
        ln_kernel<<<ROWS / 4, 128, 0, stream>>>(yd, dln_g + (size_t)(i * 3) * D,
                                                dln_b + (size_t)(i * 3) * D, hbuf);
        gemm(hbuf, sw + 0 * (size_t)D * D, sb + 0 * D, nullptr, qb, ROWS, D, D, 0);
        gemm(hbuf, sw + 1 * (size_t)D * D, sb + 1 * D, nullptr, kb, ROWS, D, D, 0);
        gemm(hbuf, sw + 2 * (size_t)D * D, sb + 2 * D, nullptr, vb, ROWS, D, D, 0);
        attn_kernel<true><<<agrid, 128, 0, stream>>>(qb, kb, vb, nullptr, ab, T, T, D);
        gemm(ab, sw + 3 * (size_t)D * D, sb + 3 * D, yd, yd, ROWS, D, D, 2);
        // cross-attention
        ln_kernel<<<ROWS / 4, 128, 0, stream>>>(yd, dln_g + (size_t)(i * 3 + 1) * D,
                                                dln_b + (size_t)(i * 3 + 1) * D, hbuf);
        gemm(hbuf, cw + 0 * (size_t)D * D, cb + 0 * D, nullptr, qb, ROWS, D, D, 0);
        gemm(enc,  cw + 1 * (size_t)D * D, cb + 1 * D, nullptr, kb, ROWS, D, D, 0);
        gemm(enc,  cw + 2 * (size_t)D * D, cb + 2 * D, nullptr, vb, ROWS, D, D, 0);
        attn_kernel<false><<<agrid, 128, 0, stream>>>(qb, kb, vb, src_mask, ab, T, S, D);
        gemm(ab, cw + 3 * (size_t)D * D, cb + 3 * D, yd, yd, ROWS, D, D, 2);
        // FFN
        ln_kernel<<<ROWS / 4, 128, 0, stream>>>(yd, dln_g + (size_t)(i * 3 + 2) * D,
                                                dln_b + (size_t)(i * 3 + 2) * D, hbuf);
        gemm(hbuf, df_w1 + (size_t)i * D * F, df_b1 + (size_t)i * F, nullptr, fb, ROWS, F, D, 1);
        gemm(fb, df_w2 + (size_t)i * F * D, df_b2 + (size_t)i * D, yd, yd, ROWS, D, F, 2);
    }
    ln_kernel<<<ROWS / 4, 128, 0, stream>>>(yd, dn_g, dn_b, hbuf);

    // ---------------- output projection + log_softmax ----------------
    gemm(hbuf, pw, pb, nullptr, out, ROWS, 32000, D, 0);
    lsm_kernel<<<ROWS, 256, 0, stream>>>(out, 32000);
}
